// FeaturePropagation_438086664391
// MI455X (gfx1250) — matrix operand order repaired and checked by reading, not compiled, not run on hardware
//
#include <hip/hip_runtime.h>

// ---------------------------------------------------------------------------
// FeaturePropagation (PointNet++ FP layer) for MI455X / gfx1250.
// B=8, N=8192, S=2048, D1=D2=256, Cin=512. Output: [B,256,N] f32.
// GEMMs run on v_wmma_f32_16x16x32_bf16 (bf16 in, f32 accumulate).
// ---------------------------------------------------------------------------

#define BATCH 8
#define NPTS  8192
#define SPTS  2048
#define DCH   256
#define CIN   512

typedef __bf16 bf16_t;
typedef bf16_t v16bf __attribute__((ext_vector_type(16)));
typedef bf16_t v8bf  __attribute__((ext_vector_type(8)));
typedef float  v8f   __attribute__((ext_vector_type(8)));

__device__ __forceinline__ unsigned short f2bf(float f) {
    unsigned int u = __float_as_uint(f);
    unsigned int r = u + 0x7FFFu + ((u >> 16) & 1u);   // round-to-nearest-even
    return (unsigned short)(r >> 16);
}

// ---------------------------------------------------------------------------
// Zero a float region (stats accumulators must start at 0 each call).
// ---------------------------------------------------------------------------
__global__ void zero_f32(float* __restrict__ p, int n) {
    for (int i = blockIdx.x * blockDim.x + threadIdx.x; i < n;
         i += gridDim.x * blockDim.x)
        p[i] = 0.0f;
}

// ---------------------------------------------------------------------------
// f32 -> bf16 weight conversion.
// ---------------------------------------------------------------------------
__global__ void to_bf16(const float* __restrict__ src,
                        unsigned short* __restrict__ dst, int n) {
    for (int i = blockIdx.x * blockDim.x + threadIdx.x; i < n;
         i += gridDim.x * blockDim.x)
        dst[i] = f2bf(src[i]);
}

// ---------------------------------------------------------------------------
// 3-NN search: for each (b,n) scan all S=2048 xyz2 points (staged in LDS),
// keep the 3 smallest squared distances (branchless insertion), emit
// normalized inverse-distance weights.
// grid: (N/128, B), block: 128
// ---------------------------------------------------------------------------
__global__ __launch_bounds__(128) void knn3(
    const float* __restrict__ xyz1, const float* __restrict__ xyz2,
    int* __restrict__ idxb, float* __restrict__ wgtb) {
    __shared__ float sx[SPTS], sy[SPTS], sz[SPTS];
    const int b = blockIdx.y;
    const int t = threadIdx.x;
    const float* src = xyz2 + (size_t)b * SPTS * 3;
    for (int i = t; i < SPTS; i += 128) {
        sx[i] = src[i * 3 + 0];
        sy[i] = src[i * 3 + 1];
        sz[i] = src[i * 3 + 2];
    }
    __syncthreads();

    const int n = blockIdx.x * 128 + t;
    const float* p = xyz1 + ((size_t)(b * NPTS + n)) * 3;
    const float px = p[0], py = p[1], pz = p[2];

    float d0 = 3.4e38f, d1 = 3.4e38f, d2 = 3.4e38f;
    int   i0 = 0, i1 = 0, i2 = 0;
    for (int s = 0; s < SPTS; ++s) {
        float dx = px - sx[s], dy = py - sy[s], dz = pz - sz[s];
        float d  = fmaf(dx, dx, fmaf(dy, dy, dz * dz));
        bool c0 = d < d0, c1 = d < d1, c2 = d < d2;
        float nd2 = c1 ? d1 : (c2 ? d : d2);
        int   ni2 = c1 ? i1 : (c2 ? s : i2);
        float nd1 = c0 ? d0 : (c1 ? d : d1);
        int   ni1 = c0 ? i0 : (c1 ? s : i1);
        d0 = c0 ? d : d0;  i0 = c0 ? s : i0;
        d1 = nd1; i1 = ni1; d2 = nd2; i2 = ni2;
    }
    float r0 = 1.0f / (d0 + 1e-8f);
    float r1 = 1.0f / (d1 + 1e-8f);
    float r2 = 1.0f / (d2 + 1e-8f);
    float inv = 1.0f / (r0 + r1 + r2);
    size_t o = ((size_t)(b * NPTS + n)) * 3;
    idxb[o] = i0; idxb[o + 1] = i1; idxb[o + 2] = i2;
    wgtb[o] = r0 * inv; wgtb[o + 1] = r1 * inv; wgtb[o + 2] = r2 * inv;
}

// ---------------------------------------------------------------------------
// Build Xcat bf16 [B][N][512]: cols 0..255 = points1, 256..511 = 3-NN
// interpolation of points2. Each thread owns a 64-byte contiguous chunk of
// one row -> fully coalesced packed stores (the transpose scatter lands on
// the read side, which stays line-efficient across the block).
// grid: (N/16, B), block: 256  (16 n-rows x 16 segments of 32 channels)
// ---------------------------------------------------------------------------
__global__ __launch_bounds__(256) void build_xcat(
    const float* __restrict__ points1, const float* __restrict__ points2,
    const int* __restrict__ idxb, const float* __restrict__ wgtb,
    unsigned short* __restrict__ Xcat) {
    const int b   = blockIdx.y;
    const int t   = threadIdx.x;
    const int n   = blockIdx.x * 16 + (t >> 4);
    const int seg = t & 15;

    const size_t o3 = ((size_t)(b * NPTS + n)) * 3;
    const int   j0 = idxb[o3], j1 = idxb[o3 + 1], j2 = idxb[o3 + 2];
    const float w0 = wgtb[o3], w1 = wgtb[o3 + 1], w2 = wgtb[o3 + 2];

    alignas(16) unsigned short tmp[32];
    if (seg < 8) {                        // copy of points1 channels
        const int c0 = seg * 32;
        const float* p1 = points1 + ((size_t)(b * DCH + c0)) * NPTS + n;
        #pragma unroll
        for (int j = 0; j < 32; ++j) tmp[j] = f2bf(p1[(size_t)j * NPTS]);
    } else {                              // interpolated points2 channels
        const int c0 = (seg - 8) * 32;
        const float* p2 = points2 + ((size_t)(b * DCH + c0)) * SPTS;
        #pragma unroll
        for (int j = 0; j < 32; ++j) {
            const float* r = p2 + (size_t)j * SPTS;
            tmp[j] = f2bf(fmaf(w0, r[j0], fmaf(w1, r[j1], w2 * r[j2])));
        }
    }
    uint4* dst = reinterpret_cast<uint4*>(
        Xcat + ((size_t)(b * NPTS + n)) * CIN + seg * 32);
    const uint4* s4 = reinterpret_cast<const uint4*>(tmp);
    dst[0] = s4[0]; dst[1] = s4[1]; dst[2] = s4[2]; dst[3] = s4[3];
}

// ---------------------------------------------------------------------------
// bf16 WMMA GEMM:  Y[b][m][n] = sum_k Wb[m][k] * Xin[b][n][k] + bias[m]
//   Xin : bf16 [B][N][K]  (K contiguous -> B-fragment = one 32B load/lane)
//   Wb  : bf16 [256][K]   (staged per 32-K slab into 16KB of LDS, shared by
//                          all 8 waves -> broadcast ds reads)
// One wave computes the full M=256 strip for 16 columns: 16 accumulator
// tiles (v8f each), 16 v_wmma_f32_16x16x32_bf16 per K step.
// Fragment layouts (A/B K order is this code's contiguous choice; C/D per CDNA5 ISA 7.12.2):
//   A 16x32 : lane row = lane&15; halves 0..15 -> K=(lane>>4)*16 (contig)
//   B 32x16 : lane col = lane&15; halves 0..15 -> K=(lane>>4)*16 (contig)
//   C/D     : element (m = (lane>>4)*8 + j, n = lane&15) in vgpr j
// grid: (N/128, B), block: 256 (8 waves, 16 columns each)
// ---------------------------------------------------------------------------
__global__ __launch_bounds__(256) void gemm_bf16(
    const unsigned short* __restrict__ Xin,
    const unsigned short* __restrict__ Wb,
    const float* __restrict__ bias,
    float* __restrict__ Y, int K) {
    __shared__ unsigned short ldsA[256 * 32];   // [m][32 k-halves], 16KB

    const int b     = blockIdx.y;
    const int n0    = blockIdx.x * 128;
    const int t     = threadIdx.x;
    const int wave  = t >> 5;
    const int lane  = t & 31;
    const int lhalf = lane >> 4;   // 0 or 1
    const int l16   = lane & 15;
    const int nw    = n0 + wave * 16 + l16;     // this lane's output column

    const unsigned short* xrow = Xin + ((size_t)(b * NPTS + nw)) * K;

    v8f acc[16] = {};
    const int ksteps = K >> 5;
    for (int ks = 0; ks < ksteps; ++ks) {
        const int k0 = ks << 5;
        __syncthreads();
        {   // cooperative A-tile load: thread t copies 32B of weight row t
            const uint4* src = reinterpret_cast<const uint4*>(Wb + (size_t)t * K + k0);
            uint4* dst = reinterpret_cast<uint4*>(&ldsA[t * 32]);
            dst[0] = src[0]; dst[1] = src[1];
        }
        __syncthreads();

        // B fragment: 16 contiguous bf16 at K = k0 + (lane>>4)*16
        v16bf bv = *reinterpret_cast<const v16bf*>(xrow + k0 + (lhalf << 4));
        if (ks + 1 < ksteps)
            __builtin_prefetch(xrow + k0 + 32, 0, 1);   // global_prefetch next slab

        #pragma unroll
        for (int mt = 0; mt < 16; ++mt) {
            const unsigned short* ap = &ldsA[(mt * 16 + l16) * 32 + (lhalf << 4)];
            v8bf alo = *reinterpret_cast<const v8bf*>(ap);        // K chunk0
            v8bf ahi = *reinterpret_cast<const v8bf*>(ap + 8);    // K chunk0+8
            v16bf av = __builtin_shufflevector(alo, ahi,
                0, 1, 2, 3, 4, 5, 6, 7, 8, 9, 10, 11, 12, 13, 14, 15);
            acc[mt] = __builtin_amdgcn_wmma_f32_16x16x32_bf16(
                false, av, false, bv, (short)0, acc[mt], false, false);
        }
    }

    #pragma unroll
    for (int mt = 0; mt < 16; ++mt) {
        #pragma unroll
        for (int j = 0; j < 8; ++j) {
            const int m = mt * 16 + lhalf * 8 + j;
            Y[((size_t)(b * DCH + m)) * NPTS + nw] = acc[mt][j] + bias[m];
        }
    }
}

// ---------------------------------------------------------------------------
// BatchNorm stats: per-channel sum / sumsq over all B*N samples.
// grid: (256, B), block: 256; LDS tree reduce then 2 f32 global atomics.
// ---------------------------------------------------------------------------
__global__ __launch_bounds__(256) void bn_stats(
    const float* __restrict__ Y, float* __restrict__ sum,
    float* __restrict__ sumsq) {
    const int m = blockIdx.x, b = blockIdx.y, t = threadIdx.x;
    const float* row = Y + ((size_t)(b * DCH + m)) * NPTS;
    float s = 0.0f, q = 0.0f;
    for (int i = t; i < NPTS; i += 256) {
        float v = row[i];
        s += v; q = fmaf(v, v, q);
    }
    __shared__ float ss[256], qq[256];
    ss[t] = s; qq[t] = q;
    __syncthreads();
    for (int off = 128; off > 0; off >>= 1) {
        if (t < off) { ss[t] += ss[t + off]; qq[t] += qq[t + off]; }
        __syncthreads();
    }
    if (t == 0) {
        atomicAdd(&sum[m], ss[0]);
        atomicAdd(&sumsq[m], qq[0]);
    }
}

// Fold mean/var/gamma/beta into per-channel scale (a) and shift (c).
__global__ void bn_finalize(const float* __restrict__ sum,
                            const float* __restrict__ sumsq,
                            const float* __restrict__ gamma,
                            const float* __restrict__ beta,
                            float* __restrict__ a, float* __restrict__ c,
                            float invCount) {
    const int m = threadIdx.x;
    float mean = sum[m] * invCount;
    float var  = fmaf(-mean, mean, sumsq[m] * invCount);
    float s    = gamma[m] * rsqrtf(var + 1e-5f);
    a[m] = s;
    c[m] = fmaf(-mean, s, beta[m]);
}

// ---------------------------------------------------------------------------
// BN + ReLU + bf16 pack + transpose -> X2 [B][N][256] for the second GEMM.
// grid: (N/16, B), block: 256 (16 n-rows x 16 segments of 16 channels)
// ---------------------------------------------------------------------------
__global__ __launch_bounds__(256) void bn_relu_pack(
    const float* __restrict__ Y1, const float* __restrict__ a,
    const float* __restrict__ c, unsigned short* __restrict__ X2) {
    __shared__ float sa[DCH], sc[DCH];
    const int t = threadIdx.x;
    sa[t] = a[t]; sc[t] = c[t];
    __syncthreads();

    const int b   = blockIdx.y;
    const int n   = blockIdx.x * 16 + (t >> 4);
    const int seg = t & 15;

    alignas(16) unsigned short tmp[16];
    #pragma unroll
    for (int j = 0; j < 16; ++j) {
        const int m = seg * 16 + j;
        float v = Y1[((size_t)(b * DCH + m)) * NPTS + n];
        tmp[j] = f2bf(fmaxf(fmaf(sa[m], v, sc[m]), 0.0f));
    }
    uint4* dst = reinterpret_cast<uint4*>(
        X2 + ((size_t)(b * NPTS + n)) * DCH + seg * 16);
    const uint4* s4 = reinterpret_cast<const uint4*>(tmp);
    dst[0] = s4[0]; dst[1] = s4[1];
}

// Final BN + ReLU, f32 output [B][256][N] (layout identical to Y2 -> trivial).
__global__ void bn_relu_out(const float* __restrict__ Y2,
                            const float* __restrict__ a,
                            const float* __restrict__ c,
                            float* __restrict__ out, int total) {
    for (int i = blockIdx.x * blockDim.x + threadIdx.x; i < total;
         i += gridDim.x * blockDim.x) {
        const int m = (i >> 13) & 255;          // N = 8192 = 2^13
        out[i] = fmaxf(fmaf(a[m], Y2[i], c[m]), 0.0f);
    }
}

// ---------------------------------------------------------------------------
// Host-side orchestration.
// ---------------------------------------------------------------------------
extern "C" void kernel_launch(void* const* d_in, const int* in_sizes, int n_in,
                              void* d_out, int out_size, void* d_ws, size_t ws_size,
                              hipStream_t stream) {
    (void)in_sizes; (void)n_in; (void)out_size; (void)ws_size;

    const float* xyz1    = (const float*)d_in[0];
    const float* xyz2    = (const float*)d_in[1];
    const float* points1 = (const float*)d_in[2];
    const float* points2 = (const float*)d_in[3];
    const float* w1      = (const float*)d_in[4];
    const float* b1      = (const float*)d_in[5];
    const float* g1      = (const float*)d_in[6];
    const float* be1     = (const float*)d_in[7];
    const float* w2      = (const float*)d_in[8];
    const float* b2      = (const float*)d_in[9];
    const float* g2      = (const float*)d_in[10];
    const float* be2     = (const float*)d_in[11];

    // Workspace layout (256B-aligned chunks). Y2 aliases Xcat (dead by then).
    char*  ws  = (char*)d_ws;
    size_t off = 0;
    auto carve = [&](size_t bytes) { size_t o = off; off += (bytes + 255) & ~(size_t)255; return o; };

    unsigned short* Xcat = (unsigned short*)(ws + carve((size_t)BATCH * NPTS * CIN * 2)); // 64MB
    float*          Y1   = (float*)         (ws + carve((size_t)BATCH * DCH * NPTS * 4)); // 64MB
    unsigned short* X2   = (unsigned short*)(ws + carve((size_t)BATCH * NPTS * DCH * 2)); // 32MB
    unsigned short* w1b  = (unsigned short*)(ws + carve((size_t)DCH * CIN * 2));
    unsigned short* w2b  = (unsigned short*)(ws + carve((size_t)DCH * DCH * 2));
    int*            idxb = (int*)           (ws + carve((size_t)BATCH * NPTS * 3 * 4));
    float*          wgtb = (float*)         (ws + carve((size_t)BATCH * NPTS * 3 * 4));
    float*          stats = (float*)        (ws + carve(8 * DCH * 4));
    float* sum1 = stats;            float* sq1 = stats + DCH;
    float* sum2 = stats + 2 * DCH;  float* sq2 = stats + 3 * DCH;
    float* a1   = stats + 4 * DCH;  float* c1  = stats + 5 * DCH;
    float* a2   = stats + 6 * DCH;  float* c2  = stats + 7 * DCH;
    float* Y2 = (float*)Xcat;   // alias: Xcat unused after first GEMM

    const float invCnt = 1.0f / (float)(BATCH * NPTS);

    zero_f32<<<4, 256, 0, stream>>>(stats, 4 * DCH);
    to_bf16<<<512, 256, 0, stream>>>(w1, w1b, DCH * CIN);
    to_bf16<<<256, 256, 0, stream>>>(w2, w2b, DCH * DCH);

    knn3<<<dim3(NPTS / 128, BATCH), 128, 0, stream>>>(xyz1, xyz2, idxb, wgtb);
    build_xcat<<<dim3(NPTS / 16, BATCH), 256, 0, stream>>>(points1, points2,
                                                           idxb, wgtb, Xcat);

    gemm_bf16<<<dim3(NPTS / 128, BATCH), 256, 0, stream>>>(Xcat, w1b, b1, Y1, CIN);
    bn_stats<<<dim3(DCH, BATCH), 256, 0, stream>>>(Y1, sum1, sq1);
    bn_finalize<<<1, DCH, 0, stream>>>(sum1, sq1, g1, be1, a1, c1, invCnt);
    bn_relu_pack<<<dim3(NPTS / 16, BATCH), 256, 0, stream>>>(Y1, a1, c1, X2);

    gemm_bf16<<<dim3(NPTS / 128, BATCH), 256, 0, stream>>>(X2, w2b, b2, Y2, DCH);
    bn_stats<<<dim3(DCH, BATCH), 256, 0, stream>>>(Y2, sum2, sq2);
    bn_finalize<<<1, DCH, 0, stream>>>(sum2, sq2, g2, be2, a2, c2, invCnt);
    bn_relu_out<<<8192, 256, 0, stream>>>(Y2, a2, c2, (float*)d_out,
                                          BATCH * DCH * NPTS);
}
